// SelfAtten2D_36601711296639
// MI455X (gfx1250) — matrix operand order repaired and checked by reading, not compiled, not run on hardware
//
#include <hip/hip_runtime.h>

#define NB 8
#define NN 4096
#define NC 512
#define NCQ 64
#define NOC 640          // 64 (g) + 64 (f) + 512 (h)
#define QTILES (NN / 16) // 256 (proj kernel row tiles)
#define QBLK 64          // attention: query rows per block
#define KTILE 32         // attention: keys per step

// padded LDS row strides (in ushorts): V rows 64B->80B, K rows 128B->144B
#define VROW 40
#define KROW 72

#define HAS_TDM __has_builtin(__builtin_amdgcn_tensor_load_to_lds)

typedef __attribute__((ext_vector_type(16))) __bf16       v16bf;
typedef __attribute__((ext_vector_type(8)))  float        v8f;
typedef __attribute__((ext_vector_type(4)))  float        v4f;
typedef __attribute__((ext_vector_type(4)))  unsigned int v4u;
typedef __attribute__((ext_vector_type(4)))  unsigned int u32x4;
typedef __attribute__((ext_vector_type(8)))  int          i32x8;
typedef __attribute__((ext_vector_type(4)))  int          i32x4;

// ---------- helpers ----------

static __device__ __forceinline__ unsigned short f2bf(float f) {
  union { float f; unsigned u; } c; c.f = f;
  unsigned r = (c.u + 0x7FFFu + ((c.u >> 16) & 1u)) >> 16; // RNE
  return (unsigned short)r;
}
static __device__ __forceinline__ unsigned pk2bf(float a, float b) {
  return (unsigned)f2bf(a) | ((unsigned)f2bf(b) << 16);
}

static __device__ __forceinline__ v8f wmma_bf16(v16bf a, v16bf b, v8f c) {
  return __builtin_amdgcn_wmma_f32_16x16x32_bf16(
      false, a, false, b, (short)0, c, false, false);
}

// two 16B chunks -> fragment (A frag, or contiguous B frag with p1 = p0+8)
static __device__ __forceinline__ v16bf ld_bf16x8x2(const unsigned short* p0,
                                                    const unsigned short* p1) {
  union { struct { v4u a, b; } s; v16bf v; } t;
  t.s.a = *reinterpret_cast<const v4u*>(p0);
  t.s.b = *reinterpret_cast<const v4u*>(p1);
  return t.v;
}
static __device__ __forceinline__ v16bf ld_bf16x16(const unsigned short* p) {
  return ld_bf16x8x2(p, p + 8);
}

static __device__ __forceinline__ float half_max(float v) {
  v = fmaxf(v, __shfl_xor(v, 1, 32));
  v = fmaxf(v, __shfl_xor(v, 2, 32));
  v = fmaxf(v, __shfl_xor(v, 4, 32));
  v = fmaxf(v, __shfl_xor(v, 8, 32));
  return v;
}
static __device__ __forceinline__ float half_sum(float v) {
  v += __shfl_xor(v, 1, 32);
  v += __shfl_xor(v, 2, 32);
  v += __shfl_xor(v, 4, 32);
  v += __shfl_xor(v, 8, 32);
  return v;
}

#if HAS_TDM
// issue one 2D TDM load: tile (tile0 x tile1) of 2B elements from gaddr into LDS,
// with row padding for bank-conflict-free fragment reads.
static __device__ __forceinline__ void tdm_load_2d(
    unsigned lds_addr, unsigned long long gaddr, unsigned g1dw0,
    unsigned td0, unsigned td1, unsigned tile0, unsigned tile1,
    unsigned stride0) {
  u32x4 g0;
  g0[0] = 1u;                                   // count=1, user mode
  g0[1] = lds_addr;                             // LDS byte address
  g0[2] = (unsigned)gaddr;                      // global addr [31:0]
  g0[3] = (unsigned)((gaddr >> 32) & 0x01FFFFFFull) | 0x80000000u; // [56:32] | type=2
  i32x8 g1;
  g1[0] = (int)g1dw0;                           // data_size=2B, pad cfg
  g1[1] = (int)(td0 << 16);                     // tensor_dim0 [15:0]
  g1[2] = (int)((td0 >> 16) | (td1 << 16));     // tensor_dim0 [31:16] | tensor_dim1 [15:0]
  g1[3] = (int)((td1 >> 16) | (tile0 << 16));   // tensor_dim1 [31:16] | tile_dim0
  g1[4] = (int)(tile1 & 0xFFFFu);               // tile_dim1 (tile_dim2=0)
  g1[5] = (int)stride0;                         // tensor_dim0_stride [31:0]
  g1[6] = 0;
  g1[7] = 0;
  i32x4 z4 = (i32x4){0, 0, 0, 0};
#if __has_include(<hip/amd_detail/amd_gfx1250_TDM.h>)
  i32x8 z8 = (i32x8){0, 0, 0, 0, 0, 0, 0, 0};
  __builtin_amdgcn_tensor_load_to_lds(g0, g1, z4, z4, z8, 0);
#else
  __builtin_amdgcn_tensor_load_to_lds(g0, g1, z4, z4, 0);
#endif
}
// data_size=1 (2B) | pad_enable | pad_interval | pad_amount(4 dwords -> code 3)
#define G1_V (0x10000u | 0x100000u | (3u << 22) | (3u << 25)) // 16-dword rows
#define G1_K (0x10000u | 0x100000u | (4u << 22) | (3u << 25)) // 32-dword rows

static __device__ __forceinline__ void wait_tensorcnt0() {
#if __has_builtin(__builtin_amdgcn_s_wait_tensorcnt)
  __builtin_amdgcn_s_wait_tensorcnt(0);
#else
  asm volatile("s_wait_tensorcnt 0x0" ::: "memory");
#endif
}
static __device__ __forceinline__ void wait_tensorcnt2() {
#if __has_builtin(__builtin_amdgcn_s_wait_tensorcnt)
  __builtin_amdgcn_s_wait_tensorcnt(2);
#else
  asm volatile("s_wait_tensorcnt 0x2" ::: "memory");
#endif
}
#endif // HAS_TDM

// ---------- kernel 1: fuse + transpose weights to bf16 ----------
__global__ __launch_bounds__(256) void prep_kernel(
    const float* __restrict__ Wg, const float* __restrict__ bg,
    const float* __restrict__ Wf, const float* __restrict__ bfv,
    const float* __restrict__ Wh, const float* __restrict__ bh,
    unsigned short* __restrict__ WT, float* __restrict__ biasC) {
  int tid = blockIdx.x * blockDim.x + threadIdx.x;
  if (tid < NOC * NC) {
    int oc = tid >> 9, cin = tid & (NC - 1);
    float w;
    if (oc < 64)       w = Wg[cin * NCQ + oc];
    else if (oc < 128) w = Wf[cin * NCQ + (oc - 64)];
    else               w = Wh[cin * NC + (oc - 128)];
    WT[(size_t)oc * NC + cin] = f2bf(w);
  }
  if (tid < NOC) {
    biasC[tid] = (tid < 64) ? bg[tid] : (tid < 128 ? bfv[tid - 64] : bh[tid - 128]);
  }
}

// ---------- kernel 2: fused projections x@[Wg|Wf|Wh]+bias -> bf16 ----------
__global__ __launch_bounds__(128) void proj_kernel(
    const float* __restrict__ x, const unsigned short* __restrict__ WT,
    const float* __restrict__ biasC,
    unsigned short* __restrict__ xg, unsigned short* __restrict__ xfp,
    unsigned short* __restrict__ xhT) {
  const int b    = blockIdx.x / QTILES;
  const int q0   = (blockIdx.x % QTILES) * 16;
  const int lane = threadIdx.x & 31;
  const int wave = threadIdx.x >> 5;
  const int h    = lane >> 4;
  const int l15  = lane & 15;

  v8f acc[10];
#pragma unroll
  for (int t = 0; t < 10; ++t) acc[t] = (v8f){0.f,0.f,0.f,0.f,0.f,0.f,0.f,0.f};

  const float* xrow = x + (size_t)(b * NN + q0 + l15) * NC;

  for (int kb = 0; kb < 16; ++kb) {
    const int cb = kb * 32;
    // preload all 10 B fragments first so loads pipeline ahead of the WMMAs
    v16bf bw[10];
#pragma unroll
    for (int t = 0; t < 10; ++t) {
      const int oc0 = (wave * 10 + t) * 16;
      bw[t] = ld_bf16x8x2(WT + (size_t)(oc0 + l15) * NC + cb + h * 8, WT + (size_t)(oc0 + l15) * NC + cb + 16 + h * 8);
    }
    // A fragment: convert 16 fp32 -> bf16 per lane
    const v4f* p0 = reinterpret_cast<const v4f*>(xrow + cb + h * 8);
    const v4f* p1 = reinterpret_cast<const v4f*>(xrow + cb + 16 + h * 8);
    v4f f0 = p0[0], f1 = p0[1], f2 = p1[0], f3 = p1[1];
    union { unsigned u[8]; v16bf v; } af;
    af.u[0] = pk2bf(f0.x, f0.y); af.u[1] = pk2bf(f0.z, f0.w);
    af.u[2] = pk2bf(f1.x, f1.y); af.u[3] = pk2bf(f1.z, f1.w);
    af.u[4] = pk2bf(f2.x, f2.y); af.u[5] = pk2bf(f2.z, f2.w);
    af.u[6] = pk2bf(f3.x, f3.y); af.u[7] = pk2bf(f3.z, f3.w);
#pragma unroll
    for (int t = 0; t < 10; ++t) acc[t] = wmma_bf16(af.v, bw[t], acc[t]);
  }

#pragma unroll
  for (int t = 0; t < 10; ++t) {
    const int oc   = (wave * 10 + t) * 16 + l15;
    const float bs = biasC[oc];
#pragma unroll
    for (int r = 0; r < 8; ++r) {
      const int rw = q0 + r + 8 * h;
      unsigned short v = f2bf(acc[t][r] + bs);
      if (oc < 64)       xg [(size_t)(b * NN + rw) * NCQ + oc]        = v;
      else if (oc < 128) xfp[(size_t)(b * NN + rw) * NCQ + (oc - 64)] = v;
      else               xhT[((size_t)b * NC + (oc - 128)) * NN + rw] = v;
    }
  }
}

// ---------- kernel 3: flash attention, Q=64 per block, TDM-staged K/V ----------
// 512 threads = 16 waves: wave w -> q-subtile (w>>2)*16, channels (w&3)*128
__global__ __launch_bounds__(512) void attn_kernel(
    const float* __restrict__ x,
    const unsigned short* __restrict__ xg, const unsigned short* __restrict__ xfp,
    const unsigned short* __restrict__ xhT,
    const float* __restrict__ gamma, float* __restrict__ out) {
  __shared__ unsigned short kbuf[2][KTILE * KROW]; // 2 x 32 keys x (64ch +pad)
  __shared__ unsigned short vbuf[2][NC * VROW];    // 2 x 512 ch x (32 keys +pad)
  __shared__ unsigned short pbuf[16][16 * 32];     // per-wave P tile

  const int b    = blockIdx.x / (NN / QBLK);
  const int qb   = (blockIdx.x % (NN / QBLK)) * QBLK;
  const int w    = threadIdx.x >> 5;
  const int lane = threadIdx.x & 31;
  const int qh   = w >> 2;
  const int cw   = w & 3;
  const int h    = lane >> 4;
  const int l15  = lane & 15;
  const int q0   = qb + qh * 16;

  // Q fragments
  const unsigned short* xgrow = xg + (size_t)(b * NN + q0 + l15) * NCQ;
  v16bf a0 = ld_bf16x8x2(xgrow + 0  + h * 16, xgrow + 8  + h * 16);
  v16bf a1 = ld_bf16x8x2(xgrow + 32 + h * 16, xgrow + 40 + h * 16);

  v8f o[8];
#pragma unroll
  for (int t = 0; t < 8; ++t) o[t] = (v8f){0.f,0.f,0.f,0.f,0.f,0.f,0.f,0.f};
  float mrow[8], lrow[8];
#pragma unroll
  for (int r = 0; r < 8; ++r) { mrow[r] = -3.0e38f; lrow[r] = 0.f; }

  unsigned short* pb = &pbuf[w][0];
  const v8f vz = (v8f){0.f,0.f,0.f,0.f,0.f,0.f,0.f,0.f};

  const unsigned short* vg = xhT + (size_t)b * NC * NN; // V, channel-major
  const unsigned short* kg = xfp + (size_t)b * NN * NCQ;

#if HAS_TDM
  if (w == 0) { // prologue: stage tile 0
    tdm_load_2d((unsigned)(size_t)&vbuf[0][0], (unsigned long long)(size_t)(vg + 0),
                G1_V, NN, NC, KTILE, NC, NN);
    tdm_load_2d((unsigned)(size_t)&kbuf[0][0], (unsigned long long)(size_t)(kg + 0),
                G1_K, NCQ, NN, NCQ, KTILE, NCQ);
  }
#endif

  for (int kt = 0; kt < NN / KTILE; ++kt) {
    const int k0 = kt * KTILE;
    __syncthreads(); // previous iteration done reading the buffer TDM will overwrite
#if HAS_TDM
    const int par = kt & 1;
    if (w == 0) {
      if (kt + 1 < NN / KTILE) {
        const int kn = k0 + KTILE;
        tdm_load_2d((unsigned)(size_t)&vbuf[par ^ 1][0],
                    (unsigned long long)(size_t)(vg + kn), G1_V, NN, NC, KTILE, NC, NN);
        tdm_load_2d((unsigned)(size_t)&kbuf[par ^ 1][0],
                    (unsigned long long)(size_t)(kg + (size_t)kn * NCQ), G1_K,
                    NCQ, NN, NCQ, KTILE, NCQ);
        wait_tensorcnt2(); // this tile's pair has landed
      } else {
        wait_tensorcnt0();
      }
    }
#else
    const int par = 0;
    { // cooperative staging fallback (padded layout identical to TDM path)
      int idx = threadIdx.x;
      for (; idx < NC * 4; idx += 512) { // V: 512 rows x 4 x 16B
        int row = idx >> 2, q = idx & 3;
        *reinterpret_cast<v4u*>(&vbuf[0][row * VROW + q * 8]) =
            *reinterpret_cast<const v4u*>(vg + (size_t)row * NN + k0 + q * 8);
      }
      if (threadIdx.x < KTILE * 8) { // K: 32 rows x 8 x 16B
        int row = threadIdx.x >> 3, q = threadIdx.x & 7;
        *reinterpret_cast<v4u*>(&kbuf[0][row * KROW + q * 8]) =
            *reinterpret_cast<const v4u*>(kg + (size_t)(k0 + row) * NCQ + q * 8);
      }
    }
#endif
    __syncthreads(); // tile ready

    // ---- logits: s = x_g (16x64) @ x_f^T (64x32), K-frags from LDS ----
    const unsigned short* kr0 = &kbuf[par][(0  + l15) * KROW + h * 16];
    const unsigned short* kr1 = &kbuf[par][(16 + l15) * KROW + h * 16];
    v16bf b00 = ld_bf16x16(kr0);
    v16bf b01 = ld_bf16x16(kr0 + 32);
    v16bf b10 = ld_bf16x16(kr1);
    v16bf b11 = ld_bf16x16(kr1 + 32);
    v8f s0 = wmma_bf16(a1, b01, wmma_bf16(a0, b00, vz));
    v8f s1 = wmma_bf16(a1, b11, wmma_bf16(a0, b10, vz));

    // ---- online softmax ----
#pragma unroll
    for (int r = 0; r < 8; ++r) {
      float s0r = s0[r], s1r = s1[r];
      float tm   = half_max(fmaxf(s0r, s1r));
      float mnew = fmaxf(mrow[r], tm);
      float sc   = __expf(mrow[r] - mnew);
      float p0   = __expf(s0r - mnew);
      float p1   = __expf(s1r - mnew);
      float ts   = half_sum(p0 + p1);
      lrow[r] = lrow[r] * sc + ts;
      mrow[r] = mnew;
#pragma unroll
      for (int t = 0; t < 8; ++t) o[t][r] *= sc;
      pb[(r + 8 * h) * 32 + l15]      = f2bf(p0);
      pb[(r + 8 * h) * 32 + 16 + l15] = f2bf(p1);
    }
    asm volatile("s_wait_dscnt 0x0" ::: "memory"); // wave-private P tile ready

    const unsigned short* pr = pb + l15 * 32 + h * 16;
    v16bf pa = ld_bf16x8x2(pr, pr + 8);

    // ---- o += P (16x32) @ V (32x16 per channel tile), V-frags from LDS ----
#pragma unroll
    for (int t = 0; t < 8; ++t) {
      const int ch = cw * 128 + t * 16 + l15;
      v16bf bh = ld_bf16x16(&vbuf[par][ch * VROW + h * 16]);
      o[t] = wmma_bf16(pa, bh, o[t]);
    }
  }

  // ---- epilogue: normalize, gamma*o + x ----
  float inv[8];
#pragma unroll
  for (int r = 0; r < 8; ++r) inv[r] = 1.0f / lrow[r];
#pragma unroll
  for (int t = 0; t < 8; ++t) {
    const int ch = cw * 128 + t * 16 + l15;
    const float g = gamma[ch];
#pragma unroll
    for (int r = 0; r < 8; ++r) {
      const int row = q0 + r + 8 * h;
      const size_t idx = (size_t)(b * NN + row) * NC + ch;
      out[idx] = g * (o[t][r] * inv[r]) + x[idx];
    }
  }
}

// ---------- host launch ----------
extern "C" void kernel_launch(void* const* d_in, const int* in_sizes, int n_in,
                              void* d_out, int out_size, void* d_ws, size_t ws_size,
                              hipStream_t stream) {
  const float* x     = (const float*)d_in[0];
  const float* Wg    = (const float*)d_in[1];
  const float* bg    = (const float*)d_in[2];
  const float* Wf    = (const float*)d_in[3];
  const float* bfv   = (const float*)d_in[4];
  const float* Wh    = (const float*)d_in[5];
  const float* bh    = (const float*)d_in[6];
  const float* gamma = (const float*)d_in[7];
  float* out = (float*)d_out;

  char* ws = (char*)d_ws;
  unsigned short* WT    = (unsigned short*)(ws);                       // 640 KB
  float*          biasC = (float*)(ws + (size_t)(1u << 20));
  unsigned short* xg    = (unsigned short*)(ws + (size_t)(2u  << 20)); // 4 MB
  unsigned short* xfp   = (unsigned short*)(ws + (size_t)(6u  << 20)); // 4 MB
  unsigned short* xhT   = (unsigned short*)(ws + (size_t)(10u << 20)); // 32 MB

  prep_kernel<<<(NOC * NC + 255) / 256, 256, 0, stream>>>(Wg, bg, Wf, bfv, Wh, bh, WT, biasC);
  proj_kernel<<<NB * QTILES, 128, 0, stream>>>(x, WT, biasC, xg, xfp, xhT);
  attn_kernel<<<NB * (NN / QBLK), 512, 0, stream>>>(x, xg, xfp, xhT, gamma, out);
}
